// IEViT21_22754736734996
// MI455X (gfx1250) — matrix operand order repaired and checked by reading, not compiled, not run on hardware
//
#include <hip/hip_runtime.h>
#include <hip/hip_bf16.h>
#include <math.h>

#define BATCH 32
#define CCH 3
#define IMGSZ 224
#define PSZ 32
#define GRD 7
#define NPATCH 49
#define EMB 768
#define NHEAD 12
#define DHEAD 64
#define LAYERS 12
#define DFF 3072
#define SMAX 62
#define MLPDIM 2048
#define NCLS 5
#define FLATDIM (SMAX * EMB)
#define KPATCH (CCH * PSZ * PSZ) /* 3072 */

typedef __bf16 v16bf __attribute__((ext_vector_type(16)));
typedef float  v8f   __attribute__((ext_vector_type(8)));

// ---------------------------------------------------------------------------
// Tiled WMMA GEMM:  C[M,N] = act( A[M,K] @ B[N,K]^T + bias )
// A, B fp32 row-major (ldA = ldB = K); converted to bf16 (hardware cvt) on the
// way into double-buffered LDS. act: 0 = none, 1 = relu, 2 = exact gelu.
// Requires K % 32 == 0. block = 128 threads (4 waves), tile 64x64, K-step 32.
// ---------------------------------------------------------------------------
#define TM 64
#define TN 64
#define TK 32
#define LDK 40

__global__ __launch_bounds__(128) void gemm_bf16_wmma(
    const float* __restrict__ A, const float* __restrict__ B,
    const float* __restrict__ bias, float* __restrict__ C,
    int M, int N, int K, int act)
{
  __shared__ __bf16 As[2][TM][LDK];
  __shared__ __bf16 Bs[2][TN][LDK];

  const int t    = threadIdx.x;
  const int lane = t & 31;
  const int wave = t >> 5;            // 0..3
  const int wm   = (wave >> 1) * 32;  // wave row offset in tile
  const int wn   = (wave & 1) * 32;   // wave col offset in tile

  const int mBase = blockIdx.y * TM;
  const int nBase = blockIdx.x * TN;

  v8f c00 = {}, c01 = {}, c10 = {}, c11 = {};

  // one thread stages one 16-float segment of one A row and one B row
  const int lr = t >> 1;        // 0..63 : tile row handled by this thread
  const int lc = (t & 1) * 16;  // 0 or 16 : k-segment
  const int grA = mBase + lr;
  const int gnB = nBase + lr;
  const bool aValid = grA < M;
  const bool bValid = gnB < N;
  const float* aSrc = A + (size_t)grA * K + lc;
  const float* bSrc = B + (size_t)gnB * K + lc;

  float ar[16], br[16];
  auto load_regs = [&](int kt) {
    if (aValid) {
      const float4* s = (const float4*)(aSrc + kt);
      #pragma unroll
      for (int q = 0; q < 4; ++q) {
        float4 f = s[q];
        ar[q * 4 + 0] = f.x; ar[q * 4 + 1] = f.y;
        ar[q * 4 + 2] = f.z; ar[q * 4 + 3] = f.w;
      }
    } else {
      #pragma unroll
      for (int q = 0; q < 16; ++q) ar[q] = 0.0f;
    }
    if (bValid) {
      const float4* s = (const float4*)(bSrc + kt);
      #pragma unroll
      for (int q = 0; q < 4; ++q) {
        float4 f = s[q];
        br[q * 4 + 0] = f.x; br[q * 4 + 1] = f.y;
        br[q * 4 + 2] = f.z; br[q * 4 + 3] = f.w;
      }
    } else {
      #pragma unroll
      for (int q = 0; q < 16; ++q) br[q] = 0.0f;
    }
  };
  auto store_lds = [&](int buf) {
    #pragma unroll
    for (int q = 0; q < 16; ++q) As[buf][lr][lc + q] = (__bf16)ar[q];
    #pragma unroll
    for (int q = 0; q < 16; ++q) Bs[buf][lr][lc + q] = (__bf16)br[q];
  };

  // fragment coordinates per CDNA5 wave32 WMMA layouts
  const int mfr = lane & 15;
  const int kbA = (lane < 16) ? 0 : 8;   // A 16x32: lanes0-15 K0..7/16..23, lanes16-31 K8..15/24..31
  const int kbB = (lane < 16) ? 0 : 8;   // B 32x16: lanes0-15 K0..7/16..23, lanes16-31 K8..15/24..31
  auto compute = [&](int buf) {
    v16bf a0, a1, b0, b1;
    #pragma unroll
    for (int j = 0; j < 8; ++j) {
      a0[j]     = As[buf][wm + mfr][kbA + j];
      a0[j + 8] = As[buf][wm + mfr][kbA + 16 + j];
      a1[j]     = As[buf][wm + 16 + mfr][kbA + j];
      a1[j + 8] = As[buf][wm + 16 + mfr][kbA + 16 + j];
    }
    #pragma unroll
    for (int j = 0; j < 16; ++j) {
      b0[j] = Bs[buf][wn + mfr][kbB + (j & 7) + ((j >> 3) << 4)];
      b1[j] = Bs[buf][wn + 16 + mfr][kbB + (j & 7) + ((j >> 3) << 4)];
    }
    c00 = __builtin_amdgcn_wmma_f32_16x16x32_bf16(false, a0, false, b0, (short)0, c00, false, false);
    c01 = __builtin_amdgcn_wmma_f32_16x16x32_bf16(false, a0, false, b1, (short)0, c01, false, false);
    c10 = __builtin_amdgcn_wmma_f32_16x16x32_bf16(false, a1, false, b0, (short)0, c10, false, false);
    c11 = __builtin_amdgcn_wmma_f32_16x16x32_bf16(false, a1, false, b1, (short)0, c11, false, false);
  };

  // software pipeline: stage k+1 in VGPRs while WMMAs consume tile k from LDS
  load_regs(0);
  store_lds(0);
  __syncthreads();
  int buf = 0;
  for (int kt = 0; kt < K; kt += TK) {
    const bool more = (kt + TK) < K;
    if (kt + 2 * TK < K) {
      if (aValid) __builtin_prefetch(aSrc + kt + 2 * TK, 0, 1);
      if (bValid) __builtin_prefetch(bSrc + kt + 2 * TK, 0, 1);
    }
    if (more) load_regs(kt + TK);
    compute(buf);
    if (more) {
      store_lds(buf ^ 1);
      __syncthreads();
      buf ^= 1;
    }
  }

  // --- epilogue: C/D layout VGPR r -> M = r (lanes0-15) or r+8 (lanes16-31), N = lane%16 ---
  const int colIn = lane & 15;
  const int rOff  = (lane < 16) ? 0 : 8;
  auto store_acc = [&](const v8f& acc, int r0, int c0) {
    int col = c0 + colIn;
    if (col >= N) return;
    float bv = bias ? bias[col] : 0.0f;
    #pragma unroll
    for (int r = 0; r < 8; ++r) {
      int row = r0 + rOff + r;
      if (row < M) {
        float v = acc[r] + bv;
        if (act == 1)      v = fmaxf(v, 0.0f);
        else if (act == 2) v = 0.5f * v * (1.0f + erff(v * 0.70710678118654752f));
        C[(size_t)row * N + col] = v;
      }
    }
  };
  store_acc(c00, mBase + wm,      nBase + wn);
  store_acc(c01, mBase + wm,      nBase + wn + 16);
  store_acc(c10, mBase + wm + 16, nBase + wn);
  store_acc(c11, mBase + wm + 16, nBase + wn + 16);
}

// ---------------------------------------------------------------------------
// Aux kernels
// ---------------------------------------------------------------------------
__global__ __launch_bounds__(256) void pool_kernel(const float* __restrict__ x,
                                                   float* __restrict__ pooled)
{
  int bc = blockIdx.x; // b*C + c
  const float* p = x + (size_t)bc * IMGSZ * IMGSZ;
  __shared__ float r[256];
  float s = 0.0f;
  for (int i = threadIdx.x; i < IMGSZ * IMGSZ; i += 256) s += p[i];
  r[threadIdx.x] = s; __syncthreads();
  for (int w = 128; w > 0; w >>= 1) {
    if (threadIdx.x < w) r[threadIdx.x] += r[threadIdx.x + w];
    __syncthreads();
  }
  if (threadIdx.x == 0) pooled[bc] = r[0] * (1.0f / (IMGSZ * IMGSZ));
}

__global__ __launch_bounds__(256) void backbone_kernel(
    const float* __restrict__ pooled, const float* __restrict__ Wbb,
    const float* __restrict__ bbb, float* __restrict__ ximg)
{
  int i = blockIdx.x * 256 + threadIdx.x;
  if (i >= BATCH * EMB) return;
  int b = i / EMB, e = i % EMB;
  float acc = bbb[e];
  #pragma unroll
  for (int c = 0; c < CCH; ++c) acc += pooled[b * CCH + c] * Wbb[c * EMB + e];
  ximg[i] = acc;
}

__global__ __launch_bounds__(256) void im2col_kernel(const float* __restrict__ x,
                                                     float* __restrict__ Ap)
{
  int i = blockIdx.x * 256 + threadIdx.x;
  if (i >= BATCH * NPATCH * KPATCH) return;
  int k  = i % KPATCH;
  int rn = i / KPATCH;
  int b  = rn / NPATCH;
  int n  = rn % NPATCH;
  int gi = n / GRD, gj = n % GRD;
  int c  = k / (PSZ * PSZ);
  int pq = k % (PSZ * PSZ);
  int p  = pq / PSZ, q = pq % PSZ;
  Ap[i] = x[(((size_t)b * CCH + c) * IMGSZ + (gi * PSZ + p)) * IMGSZ + (gj * PSZ + q)];
}

// h right-aligned: rows 0..11 zero, row 12 = cls + pos[0], rows 13..61 = patches + bp + pos[1+n]
__global__ __launch_bounds__(256) void init_h_kernel(
    const float* __restrict__ patchO, const float* __restrict__ bp,
    const float* __restrict__ cls, const float* __restrict__ pe,
    float* __restrict__ h)
{
  int i = blockIdx.x * 256 + threadIdx.x;
  if (i >= BATCH * SMAX * EMB) return;
  int e  = i % EMB;
  int rs = i / EMB;
  int b  = rs / SMAX;
  int s  = rs % SMAX;
  float v = 0.0f;
  if (s == 12) {
    v = cls[e] + pe[e];
  } else if (s > 12) {
    int n = s - 13;
    v = patchO[((size_t)b * NPATCH + n) * EMB + e] + bp[e] + pe[(size_t)(n + 1) * EMB + e];
  }
  h[i] = v;
}

// per (b, head, query-row): scores over valid keys, softmax, weighted V sum
__global__ __launch_bounds__(64) void attn_kernel(const float* __restrict__ qkv,
                                                  float* __restrict__ outb, int off)
{
  int s = blockIdx.x, hh = blockIdx.y, b = blockIdx.z;
  int t = threadIdx.x; // 0..63 == DHEAD
  float* orow = outb + ((size_t)(b * SMAX + s)) * EMB + hh * DHEAD;
  if (s < off) { orow[t] = 0.0f; return; }

  __shared__ float qsm[DHEAD];
  __shared__ float sc[64];
  __shared__ float red[64];

  const float* base = qkv + (size_t)b * SMAX * 3 * EMB;
  const float* qrow = base + (size_t)s * 3 * EMB + hh * DHEAD;
  qsm[t] = qrow[t];
  __syncthreads();

  const int nk = SMAX - off;
  float score = -INFINITY;
  if (t < nk) {
    const float* krow = base + (size_t)(off + t) * 3 * EMB + EMB + hh * DHEAD;
    float acc = 0.0f;
    #pragma unroll 8
    for (int d = 0; d < DHEAD; ++d) acc += qsm[d] * krow[d];
    score = acc * 0.125f; // 1/sqrt(64)
  }
  sc[t] = score; red[t] = score;
  __syncthreads();
  for (int w = 32; w > 0; w >>= 1) {
    if (t < w) red[t] = fmaxf(red[t], red[t + w]);
    __syncthreads();
  }
  float mx = red[0];
  __syncthreads();
  float ex = (t < nk) ? expf(sc[t] - mx) : 0.0f;
  sc[t] = ex; red[t] = ex;
  __syncthreads();
  for (int w = 32; w > 0; w >>= 1) {
    if (t < w) red[t] += red[t + w];
    __syncthreads();
  }
  float inv = 1.0f / red[0];
  __syncthreads();

  float acc = 0.0f;
  for (int k = 0; k < nk; ++k) {
    const float* vrow = base + (size_t)(off + k) * 3 * EMB + 2 * EMB + hh * DHEAD;
    acc += sc[k] * vrow[t];
  }
  orow[t] = acc * inv;
}

// h[row] = LN(h[row] + add[row]) * sc + bs   (add may be null; in-place on h)
__global__ __launch_bounds__(256) void add_ln_kernel(
    float* __restrict__ h, const float* __restrict__ add,
    const float* __restrict__ sc, const float* __restrict__ bs)
{
  int row = blockIdx.x;
  int t = threadIdx.x;
  float* hp = h + (size_t)row * EMB;
  const float* ap = add ? add + (size_t)row * EMB : nullptr;
  __shared__ float r1[256], r2[256];
  float s1 = 0.0f, s2 = 0.0f;
  for (int e = t; e < EMB; e += 256) {
    float v = hp[e] + (ap ? ap[e] : 0.0f);
    s1 += v; s2 += v * v;
  }
  r1[t] = s1; r2[t] = s2; __syncthreads();
  for (int w = 128; w > 0; w >>= 1) {
    if (t < w) { r1[t] += r1[t + w]; r2[t] += r2[t + w]; }
    __syncthreads();
  }
  float mean = r1[0] * (1.0f / EMB);
  float var  = r2[0] * (1.0f / EMB) - mean * mean;
  float inv  = rsqrtf(var + 1e-5f);
  __syncthreads();
  for (int e = t; e < EMB; e += 256) {
    float v = hp[e] + (ap ? ap[e] : 0.0f);
    hp[e] = (v - mean) * inv * sc[e] + bs[e];
  }
}

__global__ __launch_bounds__(256) void prepend_kernel(const float* __restrict__ ximg,
                                                      float* __restrict__ h, int off)
{
  int i = blockIdx.x * 256 + threadIdx.x;
  if (i >= BATCH * EMB) return;
  int b = i / EMB, e = i % EMB;
  h[((size_t)b * SMAX + (off - 1)) * EMB + e] = ximg[i];
}

__global__ __launch_bounds__(256) void head2_kernel(
    const float* __restrict__ mid, const float* __restrict__ W,
    const float* __restrict__ bias, float* __restrict__ out)
{
  int c = blockIdx.x, b = blockIdx.y;
  __shared__ float r[256];
  const float* a = mid + (size_t)b * MLPDIM;
  const float* w = W + (size_t)c * MLPDIM;
  float s = 0.0f;
  for (int k = threadIdx.x; k < MLPDIM; k += 256) s += a[k] * w[k];
  r[threadIdx.x] = s; __syncthreads();
  for (int w2 = 128; w2 > 0; w2 >>= 1) {
    if (threadIdx.x < w2) r[threadIdx.x] += r[threadIdx.x + w2];
    __syncthreads();
  }
  if (threadIdx.x == 0) out[(size_t)b * NCLS + c] = r[0] + bias[c];
}

// ---------------------------------------------------------------------------
extern "C" void kernel_launch(void* const* d_in, const int* in_sizes, int n_in,
                              void* d_out, int out_size, void* d_ws, size_t ws_size,
                              hipStream_t stream)
{
  (void)in_sizes; (void)n_in; (void)out_size; (void)ws_size;
  const float* x    = (const float*)d_in[0];
  const float* Wp   = (const float*)d_in[1];
  const float* bp   = (const float*)d_in[2];
  const float* cls  = (const float*)d_in[3];
  const float* pe   = (const float*)d_in[4];   // (E,8,8) reinterpreted flat as [64,768]
  const float* Wbb  = (const float*)d_in[5];
  const float* bbb  = (const float*)d_in[6];
  const float* Wqkv = (const float*)d_in[7];
  const float* bqkv = (const float*)d_in[8];
  const float* Wo   = (const float*)d_in[9];
  const float* bo   = (const float*)d_in[10];
  const float* ln1s = (const float*)d_in[11];
  const float* ln1b = (const float*)d_in[12];
  const float* W1   = (const float*)d_in[13];
  const float* b1   = (const float*)d_in[14];
  const float* W2   = (const float*)d_in[15];
  const float* b2   = (const float*)d_in[16];
  const float* ln2s = (const float*)d_in[17];
  const float* ln2b = (const float*)d_in[18];
  const float* lnfs = (const float*)d_in[19];
  const float* lnfb = (const float*)d_in[20];
  const float* Wh1  = (const float*)d_in[21];
  const float* bh1  = (const float*)d_in[22];
  const float* Wh2  = (const float*)d_in[23];
  const float* bh2  = (const float*)d_in[24];
  float* out = (float*)d_out;

  float* ws = (float*)d_ws;
  size_t woff = 0;
  auto alloc = [&](size_t n) { float* p = ws + woff; woff += n; return p; };
  float* h      = alloc((size_t)BATCH * SMAX * EMB);       // [32,62,768] right-aligned
  float* tmp    = alloc((size_t)BATCH * SMAX * EMB);
  float* qkvb   = alloc((size_t)BATCH * SMAX * 3 * EMB);
  float* ffb    = alloc((size_t)BATCH * SMAX * DFF);
  float* pA     = alloc((size_t)BATCH * NPATCH * KPATCH);  // im2col
  float* pO     = alloc((size_t)BATCH * NPATCH * EMB);
  float* pooled = alloc(BATCH * CCH);
  float* ximg   = alloc(BATCH * EMB);
  float* mid    = alloc(BATCH * MLPDIM);

  // backbone stub: global avg pool + linear
  pool_kernel<<<BATCH * CCH, 256, 0, stream>>>(x, pooled);
  backbone_kernel<<<(BATCH * EMB + 255) / 256, 256, 0, stream>>>(pooled, Wbb, bbb, ximg);

  // patch embedding = im2col + GEMM (Wp is naturally [768, 3072] row-major)
  im2col_kernel<<<(BATCH * NPATCH * KPATCH + 255) / 256, 256, 0, stream>>>(x, pA);
  {
    dim3 g((EMB + TN - 1) / TN, (BATCH * NPATCH + TM - 1) / TM);
    gemm_bf16_wmma<<<g, 128, 0, stream>>>(pA, Wp, nullptr, pO,
                                          BATCH * NPATCH, EMB, KPATCH, 0);
  }
  init_h_kernel<<<(BATCH * SMAX * EMB + 255) / 256, 256, 0, stream>>>(pO, bp, cls, pe, h);

  const int M = BATCH * SMAX; // 1984 rows, GEMMs run over the full buffer
  for (int i = 0; i < LAYERS; ++i) {
    const int off = 12 - i; // valid rows are [off, 62)
    { dim3 g((3 * EMB + TN - 1) / TN, (M + TM - 1) / TM);
      gemm_bf16_wmma<<<g, 128, 0, stream>>>(h, Wqkv + (size_t)i * 3 * EMB * EMB,
                                            bqkv + (size_t)i * 3 * EMB, qkvb,
                                            M, 3 * EMB, EMB, 0); }
    { dim3 g(SMAX, NHEAD, BATCH);
      attn_kernel<<<g, DHEAD, 0, stream>>>(qkvb, tmp, off); }
    { dim3 g((EMB + TN - 1) / TN, (M + TM - 1) / TM);
      gemm_bf16_wmma<<<g, 128, 0, stream>>>(tmp, Wo + (size_t)i * EMB * EMB,
                                            bo + (size_t)i * EMB, ffb,
                                            M, EMB, EMB, 0); }
    add_ln_kernel<<<M, 256, 0, stream>>>(h, ffb, ln1s + (size_t)i * EMB, ln1b + (size_t)i * EMB);
    { dim3 g((DFF + TN - 1) / TN, (M + TM - 1) / TM);
      gemm_bf16_wmma<<<g, 128, 0, stream>>>(h, W1 + (size_t)i * DFF * EMB,
                                            b1 + (size_t)i * DFF, ffb,
                                            M, DFF, EMB, 1); } // fused ReLU
    { dim3 g((EMB + TN - 1) / TN, (M + TM - 1) / TM);
      gemm_bf16_wmma<<<g, 128, 0, stream>>>(ffb, W2 + (size_t)i * EMB * DFF,
                                            b2 + (size_t)i * EMB, tmp,
                                            M, EMB, DFF, 0); }
    add_ln_kernel<<<M, 256, 0, stream>>>(h, tmp, ln2s + (size_t)i * EMB, ln2b + (size_t)i * EMB);
    prepend_kernel<<<(BATCH * EMB + 255) / 256, 256, 0, stream>>>(ximg, h, off);
  }

  // final LN, then h is directly the [32, 47616] flat matrix
  add_ln_kernel<<<M, 256, 0, stream>>>(h, nullptr, lnfs, lnfb);
  { dim3 g((MLPDIM + TN - 1) / TN, (BATCH + TM - 1) / TM);
    gemm_bf16_wmma<<<g, 128, 0, stream>>>(h, Wh1, bh1, mid,
                                          BATCH, MLPDIM, FLATDIM, 2); } // fused exact GELU
  { dim3 g(NCLS, BATCH);
    head2_kernel<<<g, 256, 0, stream>>>(mid, Wh2, bh2, out); }
}
